// MultiHeadLatentAttention_32615981646513
// MI455X (gfx1250) — compile-verified, matrix operand order checked by reading, not run on hardware
//
#include <hip/hip_runtime.h>
#include <cstddef>

// ---------------------------------------------------------------------------
// Types for CDNA5 WMMA (gfx1250, wave32)
// ---------------------------------------------------------------------------
typedef __attribute__((ext_vector_type(16))) __bf16 v16bf;
typedef __attribute__((ext_vector_type(8)))  float  v8f;

// fp32 -> bf16, round-to-nearest-even (bit math, no libcalls)
__device__ __forceinline__ unsigned short f2bf(float f) {
    unsigned u = __builtin_bit_cast(unsigned, f);
    u += 0x7FFFu + ((u >> 16) & 1u);
    return (unsigned short)(u >> 16);
}

// Wave-relative LDS byte address of a __shared__ object (generic -> as(3))
__device__ __forceinline__ unsigned lds_off(void* p) {
    return (unsigned)(unsigned long long)(__attribute__((address_space(3))) void*)p;
}

// CDNA5 async DMA: 16B/lane global -> LDS, tracked by ASYNCcnt (no VGPR data)
__device__ __forceinline__ void async_load_b128(unsigned lds, const void* g) {
    asm volatile("global_load_async_to_lds_b128 %0, %1, off"
                 :: "v"(lds), "v"((unsigned long long)(size_t)g)
                 : "memory");
}
__device__ __forceinline__ void wait_async0() {   // all DMA done
    asm volatile("s_wait_asynccnt 0x0" ::: "memory");
}
__device__ __forceinline__ void wait_async4() {   // current tile done, next in flight
    asm volatile("s_wait_asynccnt 0x4" ::: "memory");
}

// Build a 16x16x32 bf16 A/B fragment (ISA 7.12.2): lane holds two contiguous
// 8-element (16B) K runs: [half*8, +8) and [16+half*8, +8). All call sites
// guarantee 16B alignment -> lowers to 2x b128 loads.
__device__ __forceinline__ v16bf load_frag(const unsigned short* p, int r1, int r2) {
    union { v16bf v; uint4 q[2]; } u;
    u.q[0] = *reinterpret_cast<const uint4*>(p + r1);
    u.q[1] = *reinterpret_cast<const uint4*>(p + r2);
    return u.v;
}

__device__ __forceinline__ v8f wmma_bf16(v16bf a, v16bf b, v8f c) {
    return __builtin_amdgcn_wmma_f32_16x16x32_bf16(false, a, false, b,
                                                   (short)0, c, false, false);
}

// ---------------------------------------------------------------------------
// One-time operand preparation kernels
// ---------------------------------------------------------------------------
__global__ __launch_bounds__(256) void convert_bf16(
    const float* __restrict__ src, unsigned short* __restrict__ dst, int n) {
    int stride = gridDim.x * blockDim.x;
    for (int i = blockIdx.x * blockDim.x + threadIdx.x; i < n; i += stride)
        dst[i] = f2bf(src[i]);
}

// W[K][N] fp32 -> Wt[N][K] bf16 (32x32 LDS tiles, coalesced both sides)
__global__ __launch_bounds__(256) void transpose_convert_bf16(
    const float* __restrict__ W, unsigned short* __restrict__ Wt, int K, int N) {
    __shared__ float tile[32][33];
    const int k0 = blockIdx.y * 32, n0 = blockIdx.x * 32;
    const int tx = threadIdx.x, ty = threadIdx.y;   // (32, 8)
#pragma unroll
    for (int r = ty; r < 32; r += 8)
        tile[r][tx] = W[(size_t)(k0 + r) * N + (n0 + tx)];
    __syncthreads();
#pragma unroll
    for (int r = ty; r < 32; r += 8)
        Wt[(size_t)(n0 + r) * K + (k0 + tx)] = f2bf(tile[tx][r]);
}

// ---------------------------------------------------------------------------
// GEMM: C[M,N] = A[M,K] @ B[K,N]; A bf16 [M][K], B pre-transposed bf16 [N][K].
// Output optionally bf16 (Cb) and/or fp32 (Cf), scaled.
// Block: 256 threads (8 waves), tile 128x128, K-step 32.
// Double-buffered LDS staged via global_load_async_to_lds_b128; compute on
// tile i overlaps the async DMA of tile i+1 (s_wait_asynccnt 4).
// Requires: M,N multiples of 128; K multiple of 32.
// ---------------------------------------------------------------------------
#define LDS_STRIDE 40   // 32 + 8 halves pad -> conflict-free fragment reads

__global__ __launch_bounds__(256) void gemm_wmma_bf16(
    const unsigned short* __restrict__ A,   // [M][K]
    const unsigned short* __restrict__ Bt,  // [N][K]
    unsigned short* __restrict__ Cb,        // [M][N] bf16 (or null)
    float* __restrict__ Cf,                 // [M][N] fp32 (or null)
    int M, int N, int K, float scale) {
    __shared__ __align__(16) unsigned short As[2][128 * LDS_STRIDE];  // [row][k]
    __shared__ __align__(16) unsigned short Bs[2][128 * LDS_STRIDE];  // [col][k]

    const int tid  = threadIdx.x;
    const int lane = tid & 31;
    const int wid  = tid >> 5;
    const int wm   = wid & 3;    // 4 row-waves, 32 rows each
    const int wn   = wid >> 2;   // 2 col-waves, 64 cols each
    const int half = lane >> 4;
    const int ln   = lane & 15;
    const int row0 = blockIdx.y * 128;
    const int col0 = blockIdx.x * 128;

    // issue one 128x32-half tile (512 x 16B chunks, 2 chunks/thread/operand)
    auto issue_tile = [&](int kb, int buf) {
#pragma unroll
        for (int j = 0; j < 2; ++j) {
            int c = tid + j * 256;
            int r = c >> 2, ch = (c & 3) * 8;
            async_load_b128(lds_off(&As[buf][r * LDS_STRIDE + ch]),
                            A  + (size_t)(row0 + r) * K + kb + ch);
            async_load_b128(lds_off(&Bs[buf][r * LDS_STRIDE + ch]),
                            Bt + (size_t)(col0 + r) * K + kb + ch);
        }
    };

    v8f acc[2][4];
#pragma unroll
    for (int mt = 0; mt < 2; ++mt)
#pragma unroll
        for (int nt = 0; nt < 4; ++nt)
#pragma unroll
            for (int g = 0; g < 8; ++g) acc[mt][nt][g] = 0.0f;

    const int r1 = half * 8;
    const int r2 = r1 + 16;
    const int ntiles = K >> 5;

    issue_tile(0, 0);
    for (int i = 0; i < ntiles; ++i) {
        const int cur = i & 1;
        if (i + 1 < ntiles) {
            issue_tile((i + 1) << 5, cur ^ 1);   // prefetch next tile
            wait_async4();                       // tile i complete (in-order)
        } else {
            wait_async0();
        }
        __syncthreads();                         // tile i visible block-wide

        v16bf af[2], bfr[4];
#pragma unroll
        for (int mt = 0; mt < 2; ++mt)
            af[mt] = load_frag(&As[cur][(wm * 32 + mt * 16 + ln) * LDS_STRIDE], r1, r2);
#pragma unroll
        for (int nt = 0; nt < 4; ++nt)
            bfr[nt] = load_frag(&Bs[cur][(wn * 64 + nt * 16 + ln) * LDS_STRIDE], r1, r2);

#pragma unroll
        for (int mt = 0; mt < 2; ++mt)
#pragma unroll
            for (int nt = 0; nt < 4; ++nt)
                acc[mt][nt] = wmma_bf16(af[mt], bfr[nt], acc[mt][nt]);

        __syncthreads();                         // done reading buf[cur]
    }

    // C/D layout: VGPR g, lanes 0-15 -> M=g, lanes 16-31 -> M=g+8; N = lane&15
    // (null checks hoisted OUTSIDE the unrolled loops -> no per-element branch)
    if (Cb) {
#pragma unroll
        for (int mt = 0; mt < 2; ++mt)
#pragma unroll
            for (int nt = 0; nt < 4; ++nt)
#pragma unroll
                for (int g = 0; g < 8; ++g) {
                    int r = row0 + wm * 32 + mt * 16 + g + half * 8;
                    int c = col0 + wn * 64 + nt * 16 + ln;
                    Cb[(size_t)r * N + c] = f2bf(acc[mt][nt][g] * scale);
                }
    }
    if (Cf) {
#pragma unroll
        for (int mt = 0; mt < 2; ++mt)
#pragma unroll
            for (int nt = 0; nt < 4; ++nt)
#pragma unroll
                for (int g = 0; g < 8; ++g) {
                    int r = row0 + wm * 32 + mt * 16 + g + half * 8;
                    int c = col0 + wn * 64 + nt * 16 + ln;
                    Cf[(size_t)r * N + c] = acc[mt][nt][g] * scale;
                }
    }
}

// ---------------------------------------------------------------------------
// Flash attention (causal), one (batch, head, 64-query tile) per block.
// Block: 128 threads = 4 waves, each wave owns 16 query rows. All operands
// bf16 (scale pre-folded into Q). K tile staged via async DMA; V transposed
// through registers (16B global loads); online softmax in registers;
// P re-layout via per-wave LDS.
// ---------------------------------------------------------------------------
#define T_SEQ 2048
#define HDIM_ALL 2048   // 16 heads * 128
#define HEAD_DIM 128
#define K_STRIDE 136    // 128 + 8 pad
#define V_STRIDE 72     // 64 + 8 pad
#define P_STRIDE 72

__global__ __launch_bounds__(128) void mla_flash_attn(
    const unsigned short* __restrict__ Qb, const unsigned short* __restrict__ Kb,
    const unsigned short* __restrict__ Vb, unsigned short* __restrict__ Ab) {
    const int bh   = blockIdx.x;
    const int b    = bh >> 4;
    const int h    = bh & 15;
    const int q0   = blockIdx.y * 64;
    const int tid  = threadIdx.x;
    const int wave = tid >> 5;
    const int lane = tid & 31;
    const int half = lane >> 4;
    const int ln   = lane & 15;

    __shared__ __align__(16) unsigned short Ks[64 * K_STRIDE];        // [key][d]
    __shared__ __align__(16) unsigned short Vs[HEAD_DIM * V_STRIDE];  // [d][key]
    __shared__ __align__(16) unsigned short Ps[4][16 * P_STRIDE];     // per-wave P

    const int r1 = half * 8;
    const int r2 = r1 + 16;

    // ---- Q fragments: raw bf16 (scale already folded in by the Q GEMM) ----
    const int qrow = q0 + wave * 16 + ln;
    const unsigned short* qptr =
        Qb + (size_t)(b * T_SEQ + qrow) * HDIM_ALL + h * HEAD_DIM;
    v16bf qa[4];
#pragma unroll
    for (int ks = 0; ks < 4; ++ks)
        qa[ks] = load_frag(qptr, ks * 32 + r1, ks * 32 + r2);

    v8f   o[8];
    float mrun[8], lrun[8];
#pragma unroll
    for (int dt = 0; dt < 8; ++dt)
#pragma unroll
        for (int g = 0; g < 8; ++g) o[dt][g] = 0.0f;
#pragma unroll
    for (int g = 0; g < 8; ++g) { mrun[g] = -3.0e38f; lrun[g] = 0.0f; }

    const int ktiles = (q0 >> 6) + 1;   // causal: key tiles with base <= q0
    for (int kt = 0; kt < ktiles; ++kt) {
        const int k0 = kt * 64;
        __syncthreads();
        // K tile: 64 x 128 halves = 1024 x 16B chunks -> async DMA, 8/thread
        // V tile: same chunks through registers, scattered transposed to LDS
#pragma unroll
        for (int j = 0; j < 8; ++j) {
            int c = tid + j * 128;
            int kk = c >> 4, ch = (c & 15) * 8;
            size_t gofs = (size_t)(b * T_SEQ + k0 + kk) * HDIM_ALL + h * HEAD_DIM + ch;
            async_load_b128(lds_off(&Ks[kk * K_STRIDE + ch]), Kb + gofs);
            union { uint4 q; unsigned short s[8]; } v;
            v.q = *reinterpret_cast<const uint4*>(Vb + gofs);
#pragma unroll
            for (int e = 0; e < 8; ++e)
                Vs[(ch + e) * V_STRIDE + kk] = v.s[e];
        }
        wait_async0();
        __syncthreads();

        // ---- S = Q @ K^T (16 queries x 64 keys per wave) ----
        v8f s[4];
#pragma unroll
        for (int nt = 0; nt < 4; ++nt) {
#pragma unroll
            for (int g = 0; g < 8; ++g) s[nt][g] = 0.0f;
#pragma unroll
            for (int ks = 0; ks < 4; ++ks) {
                v16bf kf = load_frag(&Ks[(nt * 16 + ln) * K_STRIDE + ks * 32], r1, r2);
                s[nt] = wmma_bf16(qa[ks], kf, s[nt]);
            }
        }

        // ---- causal mask on the diagonal tile ----
        if (k0 == q0) {
#pragma unroll
            for (int nt = 0; nt < 4; ++nt)
#pragma unroll
                for (int g = 0; g < 8; ++g) {
                    int qg = wave * 16 + g + half * 8;
                    int kg = nt * 16 + ln;
                    if (kg > qg) s[nt][g] = -1.0e30f;
                }
        }

        // ---- online softmax (rows in C-layout; 16-lane xor reductions) ----
        float tmax[8];
#pragma unroll
        for (int g = 0; g < 8; ++g)
            tmax[g] = fmaxf(fmaxf(s[0][g], s[1][g]), fmaxf(s[2][g], s[3][g]));
#pragma unroll
        for (int off = 8; off >= 1; off >>= 1)
#pragma unroll
            for (int g = 0; g < 8; ++g)
                tmax[g] = fmaxf(tmax[g], __shfl_xor(tmax[g], off, 32));

        float alpha[8];
#pragma unroll
        for (int g = 0; g < 8; ++g) {
            float mn = fmaxf(mrun[g], tmax[g]);
            alpha[g] = __expf(mrun[g] - mn);
            mrun[g]  = mn;
        }
#pragma unroll
        for (int nt = 0; nt < 4; ++nt)
#pragma unroll
            for (int g = 0; g < 8; ++g)
                s[nt][g] = __expf(s[nt][g] - mrun[g]);

        float rs[8];
#pragma unroll
        for (int g = 0; g < 8; ++g)
            rs[g] = s[0][g] + s[1][g] + s[2][g] + s[3][g];
#pragma unroll
        for (int off = 8; off >= 1; off >>= 1)
#pragma unroll
            for (int g = 0; g < 8; ++g)
                rs[g] += __shfl_xor(rs[g], off, 32);

#pragma unroll
        for (int g = 0; g < 8; ++g) lrun[g] = lrun[g] * alpha[g] + rs[g];
#pragma unroll
        for (int dt = 0; dt < 8; ++dt)
#pragma unroll
            for (int g = 0; g < 8; ++g) o[dt][g] *= alpha[g];

        // ---- P (C-layout) -> per-wave LDS -> A-fragments ----
        unsigned short* pw = &Ps[wave][0];
#pragma unroll
        for (int nt = 0; nt < 4; ++nt)
#pragma unroll
            for (int g = 0; g < 8; ++g)
                pw[(g + half * 8) * P_STRIDE + nt * 16 + ln] = f2bf(s[nt][g]);
        // (same-wave DS ops are in-order; buffer is per-wave)

        // ---- O += P @ V ----
#pragma unroll
        for (int ks2 = 0; ks2 < 2; ++ks2) {
            v16bf pf = load_frag(&pw[ln * P_STRIDE + ks2 * 32], r1, r2);
#pragma unroll
            for (int dt = 0; dt < 8; ++dt) {
                v16bf vf = load_frag(&Vs[(dt * 16 + ln) * V_STRIDE + ks2 * 32], r1, r2);
                o[dt] = wmma_bf16(pf, vf, o[dt]);
            }
        }
    }

    // ---- epilogue: normalize, write bf16 attention output ----
#pragma unroll
    for (int dt = 0; dt < 8; ++dt)
#pragma unroll
        for (int g = 0; g < 8; ++g) {
            int m = g + half * 8;
            size_t gidx = (size_t)(b * T_SEQ + q0 + wave * 16 + m) * HDIM_ALL
                        + h * HEAD_DIM + dt * 16 + ln;
            Ab[gidx] = f2bf(o[dt][g] / lrun[g]);
        }
}

// ---------------------------------------------------------------------------
// Host orchestration
// ---------------------------------------------------------------------------
extern "C" void kernel_launch(void* const* d_in, const int* in_sizes, int n_in,
                              void* d_out, int out_size, void* d_ws, size_t ws_size,
                              hipStream_t stream) {
    (void)in_sizes; (void)n_in; (void)out_size; (void)ws_size;

    const float* x     = (const float*)d_in[0];   // (2, 2048, 2048)
    const float* Wq    = (const float*)d_in[1];   // (2048, 2048)
    const float* Wdown = (const float*)d_in[2];   // (2048, 512)
    const float* Wk_up = (const float*)d_in[3];   // (512, 2048)
    const float* Wv_up = (const float*)d_in[4];   // (512, 2048)
    const float* Wo    = (const float*)d_in[5];   // (2048, 2048)

    float* out   = (float*)d_out;                  // (2,2048,2048)
    float* c_out = out + (size_t)2 * 2048 * 2048;  // (2,2048,512), 2nd tuple elem

    const size_t MT = 4096;   // B*T rows
    unsigned short* p   = (unsigned short*)d_ws;
    unsigned short* xb  = p;  p += MT * 2048;            // x, bf16
    unsigned short* WqT = p;  p += (size_t)2048 * 2048;  // Wq^T   [2048][2048]
    unsigned short* WdT = p;  p += (size_t)512  * 2048;  // Wdown^T [512][2048]
    unsigned short* WkT = p;  p += (size_t)2048 * 512;   // Wk_up^T [2048][512]
    unsigned short* WvT = p;  p += (size_t)2048 * 512;   // Wv_up^T [2048][512]
    unsigned short* WoT = p;  p += (size_t)2048 * 2048;  // Wo^T   [2048][2048]
    unsigned short* Qb  = p;  p += MT * 2048;
    unsigned short* cb  = p;  p += MT * 512;
    unsigned short* Kb  = p;  p += MT * 2048;
    unsigned short* Vb  = p;  p += MT * 2048;
    unsigned short* Ab  = p;  p += MT * 2048;

    const float qscale = 0.08838834764831845f;   // 1/sqrt(128)

    // ---- one-time bf16 conversion / weight transposition ----
    convert_bf16<<<4096, 256, 0, stream>>>(x, xb, (int)(MT * 2048));
    dim3 tb(32, 8);
    transpose_convert_bf16<<<dim3(2048 / 32, 2048 / 32), tb, 0, stream>>>(Wq,    WqT, 2048, 2048);
    transpose_convert_bf16<<<dim3(512  / 32, 2048 / 32), tb, 0, stream>>>(Wdown, WdT, 2048, 512);
    transpose_convert_bf16<<<dim3(2048 / 32, 512  / 32), tb, 0, stream>>>(Wk_up, WkT, 512, 2048);
    transpose_convert_bf16<<<dim3(2048 / 32, 512  / 32), tb, 0, stream>>>(Wv_up, WvT, 512, 2048);
    transpose_convert_bf16<<<dim3(2048 / 32, 2048 / 32), tb, 0, stream>>>(Wo,    WoT, 2048, 2048);

    dim3 blk(256);
    // Q = (x @ Wq) * 1/sqrt(Dh)  -> bf16
    gemm_wmma_bf16<<<dim3(16, 32), blk, 0, stream>>>(xb, WqT, Qb, nullptr,
                                                     (int)MT, 2048, 2048, qscale);
    // c = x @ Wdown -> fp32 output slot + bf16 working copy
    gemm_wmma_bf16<<<dim3(4, 32), blk, 0, stream>>>(xb, WdT, cb, c_out,
                                                    (int)MT, 512, 2048, 1.0f);
    // K = c @ Wk_up -> bf16
    gemm_wmma_bf16<<<dim3(16, 32), blk, 0, stream>>>(cb, WkT, Kb, nullptr,
                                                     (int)MT, 2048, 512, 1.0f);
    // V = c @ Wv_up -> bf16
    gemm_wmma_bf16<<<dim3(16, 32), blk, 0, stream>>>(cb, WvT, Vb, nullptr,
                                                     (int)MT, 2048, 512, 1.0f);
    // causal SDPA: grid = (B*H, T/64)
    mla_flash_attn<<<dim3(32, 32), dim3(128), 0, stream>>>(Qb, Kb, Vb, Ab);
    // out = attn @ Wo -> fp32
    gemm_wmma_bf16<<<dim3(16, 32), blk, 0, stream>>>(Ab, WoT, nullptr, out,
                                                     (int)MT, 2048, 2048, 1.0f);
}
